// SpectralTokenizer_36103495090536
// MI455X (gfx1250) — matrix operand order repaired and checked by reading, not compiled, not run on hardware
//
#include <hip/hip_runtime.h>
#include <math.h>

// ---------------------------------------------------------------------------
// SpectralTokenizer for MI455X (gfx1250, wave32, WMMA).
//
// 64-pt Hann-windowed rfft of 5 hop-8 frames per 100-sample row,
// log1p(|spec|), freq-major flatten -> 165 f32 per row.
//
// GEMM formulation: (frame-rows x 64) x (64 x 33) twice (cos & sin basis,
// Hann window folded into the basis along K).
// WMMA: v_wmma_f32_16x16x32_f16, K=64 as two K=32 steps, N=33 as 3 tiles.
//
// Round-2 changes:
//  - B basis built into LDS by a rolled loop (256 thr x 24 halfs), each wave
//    then loads its 12 fragments via ds_load_b128 -> kills ~770 static VALU
//    of unrolled transcendentals and amortizes per-block, not per-wave.
//  - raw v_sqrt_f32 / v_log_f32 (__builtin_amdgcn_sqrtf/logf) instead of the
//    IEEE libm expansions in the epilogue.
//  - per-v store-index base hoisted out of the j loop.
// ---------------------------------------------------------------------------

typedef __attribute__((ext_vector_type(16))) _Float16 v16h;
typedef __attribute__((ext_vector_type(8)))  float    v8f;

#define N_FFT      64
#define HOP        8
#define PATCH_LEN  100
#define N_FRAMES   5
#define N_FREQ     33
#define OUT_ROW    (N_FREQ * N_FRAMES)   // 165
#define LN2        0.69314718055994530942f

// B fragment storage: [frag F][lane L][elem e], F = j*4 + q*2 + cs
// 12 frags * 32 lanes * 16 halfs = 6144 halfs = 12 KB LDS
#define N_FRAG     12
#define B_ELEMS    (N_FRAG * 32 * 16)

__global__ __launch_bounds__(256)
void stft_wmma_kernel(const float* __restrict__ patches,
                      float* __restrict__ out,
                      int nFrameRows, int nTiles)
{
    __shared__ __align__(16) _Float16 Bsh[B_ELEMS];

    const int lane = (int)(threadIdx.x & 31);
    const int hi   = lane >> 4;        // half-wave group (0/1)
    const int m    = lane & 15;        // A-row / B-col / C-col index
    const int wavesPerBlock = (int)(blockDim.x >> 5);
    const int globalWave = (int)blockIdx.x * wavesPerBlock + (int)(threadIdx.x >> 5);
    const int waveStride = (int)gridDim.x * wavesPerBlock;

    // ---- Build windowed DFT basis into LDS (rolled loop, whole block) ----
    // B (32x16 f16) layout: lane L holds column N = L&15,
    // element e holds K = e + 16*(L>>4).  Two K-chunks (q) cover K=0..63.
    //   cs=0: w[n]*cos(2*pi*n*k/64)   cs=1: w[n]*sin(2*pi*n*k/64)
    // (sign of im irrelevant: only re^2+im^2 used)
    const float PI_32 = 0.0981747704246810387f;   // pi/32
    for (int idx = (int)threadIdx.x; idx < B_ELEMS; idx += (int)blockDim.x) {
        const int e  = idx & 15;
        const int L  = (idx >> 4) & 31;
        const int F  = idx >> 9;          // 0..11
        const int cs = F & 1;
        const int q  = (F >> 1) & 1;
        const int j  = F >> 2;
        const int n  = 32 * q + 16 * (L >> 4) + e;   // sample index 0..63
        const int k  = 16 * j + (L & 15);            // freq index
        const float w   = 0.5f - 0.5f * __cosf((float)n * PI_32);
        const float ang = (float)((n * k) & 63) * PI_32;  // exact mod-64, < 2pi
        float val = cs ? (w * __sinf(ang)) : (w * __cosf(ang));
        if (k > 32) val = 0.0f;           // pad columns beyond 33 freqs
        Bsh[idx] = (_Float16)val;
    }
    __syncthreads();

    // ---- Each wave pulls its 12 fragments from LDS (ds_load_b128 x2 each)
    v16h bc[3][2], bs[3][2];
#pragma unroll
    for (int j = 0; j < 3; ++j) {
#pragma unroll
        for (int q = 0; q < 2; ++q) {
            const int Fc = j * 4 + q * 2 + 0;
            const int Fs = j * 4 + q * 2 + 1;
            bc[j][q] = *(const v16h*)&Bsh[(Fc * 32 + lane) * 16];
            bs[j][q] = *(const v16h*)&Bsh[(Fs * 32 + lane) * 16];
        }
    }

    // ---- Tile loop: 16 frame-rows per wave per iteration -----------------
    for (int tile = globalWave; tile < nTiles; tile += waveStride) {
        // ---- A fragment load: lane L holds row M = L&15 ------------------
        //   e in [0,16): K = 32q + 16*hi + e
        int g = tile * 16 + m;
        if (g >= nFrameRows) g = nFrameRows - 1;      // clamp (benign dup)
        const int row = g / N_FRAMES;                 // one div per tile
        const int fr  = g - row * N_FRAMES;
        const float* src = patches + (size_t)row * PATCH_LEN + fr * HOP;

        // Uniform-offset prefetch of next tile's region (hint only)
        if (tile + waveStride < nTiles)
            __builtin_prefetch(src + (size_t)waveStride * 320, 0, 1);

        v16h a01[2];
#pragma unroll
        for (int q = 0; q < 2; ++q) {
            const float* p = src + 32 * q + 16 * hi;  // 16B aligned
            const float4 x0 = *(const float4*)(p +  0);
            const float4 x1 = *(const float4*)(p +  4);
            const float4 x2 = *(const float4*)(p +  8);
            const float4 x3 = *(const float4*)(p + 12);
            v16h a;
            a[0]=(_Float16)x0.x;  a[1]=(_Float16)x0.y;  a[2]=(_Float16)x0.z;  a[3]=(_Float16)x0.w;
            a[4]=(_Float16)x1.x;  a[5]=(_Float16)x1.y;  a[6]=(_Float16)x1.z;  a[7]=(_Float16)x1.w;
            a[8]=(_Float16)x2.x;  a[9]=(_Float16)x2.y;  a[10]=(_Float16)x2.z; a[11]=(_Float16)x2.w;
            a[12]=(_Float16)x3.x; a[13]=(_Float16)x3.y; a[14]=(_Float16)x3.z; a[15]=(_Float16)x3.w;
            a01[q] = a;
        }

        // ---- 12 WMMAs: {cos,sin} x 3 N-tiles x 2 K-chunks ----------------
        v8f accC[3], accS[3];
#pragma unroll
        for (int j = 0; j < 3; ++j) { accC[j] = (v8f){}; accS[j] = (v8f){}; }
#pragma unroll
        for (int j = 0; j < 3; ++j) {
#pragma unroll
            for (int q = 0; q < 2; ++q) {
                accC[j] = __builtin_amdgcn_wmma_f32_16x16x32_f16(
                    false, a01[q], false, bc[j][q], (short)0, accC[j], false, false);
                accS[j] = __builtin_amdgcn_wmma_f32_16x16x32_f16(
                    false, a01[q], false, bs[j][q], (short)0, accS[j], false, false);
            }
        }

        // ---- Epilogue ----------------------------------------------------
        // C/D layout: lane L, vgpr v -> (M = v + 8*(L>>4), N = L&15).
        // Flat out index for frame-row gg = g0+v, freq k = 16j+m:
        //   idx = idx0 + 80*j + v + 160*c,  c = (f0+v>=5)+(f0+v>=10)
        const int g0 = tile * 16 + 8 * hi;
        const int r0 = g0 / N_FRAMES;                 // one div per tile
        const int f0 = g0 - r0 * N_FRAMES;
        const unsigned idx0 = (unsigned)r0 * OUT_ROW + (unsigned)m * N_FRAMES
                            + (unsigned)f0;

#pragma unroll
        for (int v = 0; v < 8; ++v) {
            if (g0 + v < nFrameRows) {
                const int fv = f0 + v;
                const unsigned c = (fv >= 5 ? 1u : 0u) + (fv >= 10 ? 1u : 0u);
                const unsigned idxv = idx0 + (unsigned)v + 160u * c;  // hoisted
#pragma unroll
                for (int j = 0; j < 3; ++j) {
                    const int k = 16 * j + m;         // freq index (per lane)
                    if (k <= 32) {                    // lane-masked, post-WMMA
                        const float re = accC[j][v];
                        const float im = accS[j][v];
                        const float mag = __builtin_amdgcn_sqrtf(re * re + im * im);
                        const float val = __builtin_amdgcn_logf(1.0f + mag) * LN2;
                        __builtin_nontemporal_store(val, out + idxv + (unsigned)(80 * j));
                    }
                }
            }
        }
    }
}

extern "C" void kernel_launch(void* const* d_in, const int* in_sizes, int n_in,
                              void* d_out, int out_size, void* d_ws, size_t ws_size,
                              hipStream_t stream) {
    const float* patches = (const float*)d_in[0];
    float* out = (float*)d_out;

    const int total      = in_sizes[0];              // B*N*100
    const int nRows      = total / PATCH_LEN;        // 262144
    const int nFrameRows = nRows * N_FRAMES;         // 1,310,720
    const int nTiles     = (nFrameRows + 15) / 16;   // 81,920

    const int threads = 256;                         // 8 waves / block
    const int wavesPerBlock = threads / 32;
    int blocks = (nTiles + wavesPerBlock - 1) / wavesPerBlock;
    if (blocks > 2048) blocks = 2048;                // grid-stride amortizes setup

    stft_wmma_kernel<<<blocks, threads, 0, stream>>>(patches, out, nFrameRows, nTiles);
}
